// QuantizeEMAReset_63866163692084
// MI455X (gfx1250) — matrix operand order repaired and checked by reading, not compiled, not run on hardware
//
#include <hip/hip_runtime.h>

typedef __attribute__((ext_vector_type(16))) _Float16 v16h;
typedef __attribute__((ext_vector_type(8)))  float    v8f;

constexpr int V = 16384;   // codebook size
constexpr int C = 128;     // code dim (K)
constexpr int N = 16;      // batch
constexpr int T = 512;     // time
constexpr int M = N * T;   // 8192 query rows
constexpr int KB = C / 32; // 4 K-blocks of 32 per WMMA chain
constexpr int RT = M / 16; // 512 row tiles
constexpr int NT = V / 16; // 1024 column (code) tiles
constexpr int NTH = NT / 2; // column tiles per V-half (512)

// ---------------------------------------------------------------------------
// Pack x (f32 [N,C,T]) into WMMA A-fragment layout (f16), 16x32 tiles.
// A 16-bit 16x32 layout (ISA 7.12.2): lane l<16 -> row M=l, halves h map to
// K = h + 8*((h>>3)+(l>>4)) within a 32-wide K block.
// Element id = ((rt*KB+kb)*32 + l)*16 + h  -> contiguous v16h per lane.
// ---------------------------------------------------------------------------
__global__ void pack_x_kernel(const float* __restrict__ x,
                              _Float16* __restrict__ packA) {
  int id = blockIdx.x * blockDim.x + threadIdx.x;   // M*C ids
  int h  = id & 15;
  int l  = (id >> 4) & 31;
  int kb = (id >> 9) & 3;
  int rt = id >> 11;
  int K  = kb * 32 + h + 8 * ((h >> 3) + (l >> 4));
  int m  = rt * 16 + (l & 15);
  int bn = m / T, t = m % T;
  // xf[m][K] = x[bn][K][t]
  packA[id] = (_Float16)x[bn * C * T + K * T + t];
}

// ---------------------------------------------------------------------------
// Pack codebook (f32 [V,C]) into WMMA B-fragment layout (f16), 32x16 tiles.
// B 16-bit 32x16 layout: lane l -> column N=(l&15), K = h + 8*((h>>3)+(l>>4)).
// ---------------------------------------------------------------------------
__global__ void pack_cb_kernel(const float* __restrict__ cb,
                               _Float16* __restrict__ packB) {
  int id = blockIdx.x * blockDim.x + threadIdx.x;   // V*C ids
  int h  = id & 15;
  int l  = (id >> 4) & 31;
  int kb = (id >> 9) & 3;
  int nt = id >> 11;
  int K   = kb * 32 + h + 8 * ((h >> 3) + (l >> 4));
  int col = nt * 16 + (l & 15);
  packB[id] = (_Float16)cb[col * C + K];
}

// |k_v|^2 in f32 (kept full precision; folded into score after WMMA)
__global__ void knorm_kernel(const float* __restrict__ cb,
                             float* __restrict__ knorm) {
  int v = blockIdx.x * blockDim.x + threadIdx.x;
  const float4* row = (const float4*)(cb + (size_t)v * C);
  float s = 0.f;
#pragma unroll
  for (int i = 0; i < C / 4; ++i) {
    float4 q = row[i];
    s += q.x * q.x + q.y * q.y + q.z * q.z + q.w * q.w;
  }
  knorm[v] = s;
}

__global__ void zero_counts_kernel(int* __restrict__ counts) {
  counts[blockIdx.x * blockDim.x + threadIdx.x] = 0;
}

// ---------------------------------------------------------------------------
// Hot kernel: 128 blocks x 8 waves = 1024 waves (2-4 waves/SIMD for latency
// hiding). Wave pair (2w, 2w+1) shares one 16-row tile of queries and splits
// the V=16384 codebook in half. Per wave: A frags in registers, loop over 512
// column tiles: 4x v_wmma_f32_16x16x32_f16 (K=128), score = |k|^2 - 2*dot,
// running argmin kept in the C/D-matrix lane layout. Final reduction:
// cross-lane min within each 16-lane half (wave32 shfl), then the wave pair
// merges through LDS.
// ---------------------------------------------------------------------------
__global__ __launch_bounds__(256) void vq_argmin_kernel(
    const _Float16* __restrict__ packA, const _Float16* __restrict__ packB,
    const float* __restrict__ knorm, int* __restrict__ code_idx,
    int* __restrict__ counts) {
  __shared__ float smv[128];  // 8 waves * 16 (row-half-major) partial mins
  __shared__ int   smi[128];

  const int lane  = threadIdx.x & 31;
  const int wave  = threadIdx.x >> 5;
  const int rt    = blockIdx.x * 4 + (wave >> 1);  // 128 blocks * 4 tiles
  const int vhalf = wave & 1;                      // which half of V
  const int vbeg  = vhalf * NTH;

  const v16h* apv = ((const v16h*)packA) + (size_t)rt * (KB * 32) + lane;
  const v16h a0 = apv[0 * 32];
  const v16h a1 = apv[1 * 32];
  const v16h a2 = apv[2 * 32];
  const v16h a3 = apv[3 * 32];

  float minv[8];
  int   mini[8];
#pragma unroll
  for (int i = 0; i < 8; ++i) { minv[i] = 3.4e38f; mini[i] = 0; }

  const int colbase = lane & 15;

  for (int vt = vbeg; vt < vbeg + NTH; ++vt) {
    const v16h* bpv = ((const v16h*)packB) + (size_t)vt * (KB * 32) + lane;
    v16h b0 = bpv[0 * 32];
    v16h b1 = bpv[1 * 32];
    v16h b2 = bpv[2 * 32];
    v16h b3 = bpv[3 * 32];
    float kn = knorm[vt * 16 + colbase];
    // global_prefetch_b8 two tiles ahead; speculative OOB prefetch is dropped
    __builtin_prefetch((const void*)(bpv + 2 * KB * 32), 0, 1);

    v8f c = {};
    c = __builtin_amdgcn_wmma_f32_16x16x32_f16(false, a0, false, b0, (short)0, c, false, false);
    c = __builtin_amdgcn_wmma_f32_16x16x32_f16(false, a1, false, b1, (short)0, c, false, false);
    c = __builtin_amdgcn_wmma_f32_16x16x32_f16(false, a2, false, b2, (short)0, c, false, false);
    c = __builtin_amdgcn_wmma_f32_16x16x32_f16(false, a3, false, b3, (short)0, c, false, false);

    const int colid = vt * 16 + colbase;
#pragma unroll
    for (int i = 0; i < 8; ++i) {
      float s = __builtin_fmaf(-2.0f, c[i], kn);   // |k|^2 - 2*x.k
      bool take = s < minv[i];                     // strict < keeps earliest id
      minv[i] = take ? s : minv[i];
      mini[i] = take ? colid : mini[i];
    }
  }

  // Reduce across the 16 lanes of each half-wave (rows 0-7 live in lanes
  // 0-15, rows 8-15 in lanes 16-31 per the 16x16 f32 C/D layout).
#pragma unroll
  for (int i = 0; i < 8; ++i) {
#pragma unroll
    for (int off = 8; off > 0; off >>= 1) {
      float ov = __shfl_xor(minv[i], off, 16);
      int   oi = __shfl_xor(mini[i], off, 16);
      bool take = (ov < minv[i]) || ((ov == minv[i]) && (oi < mini[i]));
      minv[i] = take ? ov : minv[i];
      mini[i] = take ? oi : mini[i];
    }
  }

  // Publish per-wave partials: slot = wave*16 + rowhalf*8 + i
  const int rowhalf = lane >> 4;
  if ((lane & 15) == 0) {
    int base = wave * 16 + rowhalf * 8;
#pragma unroll
    for (int i = 0; i < 8; ++i) { smv[base + i] = minv[i]; smi[base + i] = mini[i]; }
  }
  __syncthreads();

  // Even wave of each pair merges its partner's half of V and writes results.
  if (((wave & 1) == 0) && ((lane & 15) == 0)) {
    int base  = wave * 16 + rowhalf * 8;
    int pbase = base + 16;                    // partner wave (wave+1)
    int rbase = rt * 16 + rowhalf * 8;
#pragma unroll
    for (int i = 0; i < 8; ++i) {
      float v  = smv[base + i];  int idx  = smi[base + i];
      float pv = smv[pbase + i]; int pidx = smi[pbase + i];
      bool take = (pv < v) || ((pv == v) && (pidx < idx));
      if (take) { v = pv; idx = pidx; }
      code_idx[rbase + i] = idx;
      atomicAdd(&counts[idx], 1);
    }
  }
}

// x_d[bn][c][t] = codebook[code_idx[bn*T + t]][c]
__global__ void dequant_kernel(const float* __restrict__ cb,
                               const int* __restrict__ code_idx,
                               float* __restrict__ out) {
  int id = blockIdx.x * blockDim.x + threadIdx.x;  // N*C*T
  int t  = id & (T - 1);
  int c  = (id >> 9) & (C - 1);   // T = 2^9
  int bn = id >> 16;              // C*T = 2^16
  int code = code_idx[bn * T + t];
  out[id] = cb[(size_t)code * C + c];
}

// Deterministic single-block perplexity from integer histogram.
__global__ void perplexity_kernel(const int* __restrict__ counts,
                                  float* __restrict__ out) {
  __shared__ float sh[1024];
  int tid = threadIdx.x;
  float lsum = 0.f;
  for (int v = tid; v < V; v += 1024) lsum += (float)counts[v];
  sh[tid] = lsum; __syncthreads();
  for (int s = 512; s > 0; s >>= 1) {
    if (tid < s) sh[tid] += sh[tid + s];
    __syncthreads();
  }
  float total = sh[0];
  __syncthreads();
  float h = 0.f;
  for (int v = tid; v < V; v += 1024) {
    float p = (float)counts[v] / total;
    h += p * logf(p + 1e-7f);
  }
  sh[tid] = h; __syncthreads();
  for (int s = 512; s > 0; s >>= 1) {
    if (tid < s) sh[tid] += sh[tid + s];
    __syncthreads();
  }
  if (tid == 0) out[0] = expf(-sh[0]);
}

extern "C" void kernel_launch(void* const* d_in, const int* in_sizes, int n_in,
                              void* d_out, int out_size, void* d_ws, size_t ws_size,
                              hipStream_t stream) {
  const float* x  = (const float*)d_in[0];   // [N, C, T]
  const float* cb = (const float*)d_in[1];   // [V, C]
  float* out = (float*)d_out;                // [N*C*T] x_d, then [1] perplexity

  // Workspace layout (~6.2 MB total)
  char* ws = (char*)d_ws;
  _Float16* packA    = (_Float16*)(ws);                           // M*C*2 = 2 MiB
  _Float16* packB    = (_Float16*)(ws + (size_t)2 * 1024 * 1024); // V*C*2 = 4 MiB
  float*    knorm    = (float*)(ws + (size_t)6 * 1024 * 1024);    // 64 KiB
  int*      code_idx = (int*)  (ws + (size_t)6 * 1024 * 1024 + 64 * 1024); // 32 KiB
  int*      counts   = (int*)  (ws + (size_t)6 * 1024 * 1024 + 96 * 1024); // 64 KiB

  pack_x_kernel     <<<(M * C) / 256, 256, 0, stream>>>(x, packA);
  pack_cb_kernel    <<<(V * C) / 256, 256, 0, stream>>>(cb, packB);
  knorm_kernel      <<<V / 256,       256, 0, stream>>>(cb, knorm);
  zero_counts_kernel<<<V / 256,       256, 0, stream>>>(counts);

  // 128 blocks * 8 waves: 4 row tiles per block, V split across wave pairs.
  vq_argmin_kernel  <<<RT / 4,        256, 0, stream>>>(packA, packB, knorm,
                                                        code_idx, counts);

  dequant_kernel    <<<(N * C * T) / 256, 256, 0, stream>>>(cb, code_idx, out);
  perplexity_kernel <<<1, 1024, 0, stream>>>(counts, out + (size_t)N * C * T);
}
